// DualAdaptiveEncoder_34050500723135
// MI455X (gfx1250) — matrix operand order repaired and checked by reading, not compiled, not run on hardware
//
#include <hip/hip_runtime.h>
#include <hip/hip_bf16.h>
#include <math.h>

// ---------------------------------------------------------------------------
// DualAdaptiveEncoder for MI455X (gfx1250, wave32, WMMA)
// Dense GEMMs -> v_wmma_f32_16x16x32_bf16, double-buffered LDS pipeline,
//                A-tile staged via global_load_async_to_lds_b128 (ASYNCcnt).
// Edge ops -> wave-per-edge reductions + f32 global atomics.
// k-th value prune -> 4-pass radix select on monotone float keys.
// ---------------------------------------------------------------------------

#define NN      16384
#define NE      65536
#define KPRUNE  6553        // max(1, int(0.1 * 65536))
#define NEG_SLOPE 0.2f
#define BETA_CROSS 0.2f
#define LAMBDA_PATCH 0.1f
#define KEY_NEG_INF 0x007FFFFFu   // keyf(-inf)

#define ASYNC_A 1           // stage A tile with global_load_async_to_lds_b128

typedef __attribute__((ext_vector_type(16))) __bf16 v16bf;
typedef __attribute__((ext_vector_type(8)))  __bf16 v8bf;
typedef __attribute__((ext_vector_type(8)))  float  v8f;

union FragBF { v8bf h[2]; v16bf v; };

__device__ __forceinline__ float lrelu(float v) { return v > 0.f ? v : v * NEG_SLOPE; }

__device__ __forceinline__ unsigned keyf(float f) {
  unsigned u = __float_as_uint(f);
  return (u & 0x80000000u) ? ~u : (u | 0x80000000u);
}
__device__ __forceinline__ float inv_keyf(unsigned k) {
  unsigned u = (k & 0x80000000u) ? (k & 0x7FFFFFFFu) : ~k;
  return __uint_as_float(u);
}
__device__ __forceinline__ float wave_sum(float v) {
  #pragma unroll
  for (int o = 16; o; o >>= 1) v += __shfl_xor(v, o, 32);
  return v;
}

#if ASYNC_A
// Per-lane async copy of 16B global -> LDS (VFLAT GLOBAL_LOAD_ASYNC_TO_LDS_B128).
// LDS dest VGPR carries the low-32 bits of the generic shared address, which
// per ISA 10.2 aperture rules is the LDS byte address.
__device__ __forceinline__ void async_ld_b128(const void* g, void* l) {
  asm volatile("global_load_async_to_lds_b128 %0, %1, off"
               :: "v"((unsigned)(size_t)l),
                  "v"((unsigned long long)(size_t)g)
               : "memory");
}
__device__ __forceinline__ void wait_async_all() {
#if __has_builtin(__builtin_amdgcn_s_wait_asynccnt)
  __builtin_amdgcn_s_wait_asynccnt(0);
#else
  asm volatile("s_wait_asynccnt 0x0" ::: "memory");
#endif
}
#else
__device__ __forceinline__ void wait_async_all() {}
#endif

// ---------------------------------------------------------------------------
// GEMM: out[M,N] = scale * (A_bf16[M,K] @ B_bf16[K,N] + bias[N]) + add0 + add1
// Block: 256 threads (8 waves). Block tile 128x128, k-step 32, 2 LDS buffers.
// Wave grid 2x4, each wave -> 64x32 = 4x2 WMMA 16x16 tiles.
// Pipeline per k-step: compute(buf) | store+async next buf | issue loads k+2
// | wait_async | barrier  (global-load latency hidden behind compute phase)
// ---------------------------------------------------------------------------
__global__ __launch_bounds__(256)
void gemm_bf16_kernel(const __bf16* __restrict__ A, const __bf16* __restrict__ B,
                      const float* __restrict__ bias,
                      const float* __restrict__ add0, const float* __restrict__ add1,
                      float scale, float* __restrict__ out,
                      int M, int K, int Nn)
{
  __shared__ __bf16 As[2][128][40];   // [m][k], 80B rows (16B aligned)
  __shared__ __bf16 Bs[2][128][40];   // transposed: [n][k]

  const int tid   = threadIdx.x;
  const int lane  = tid & 31;
  const int wave  = tid >> 5;
  const int waveM = (wave >> 2) * 64;   // 0 / 64
  const int waveN = (wave & 3) * 32;    // 0 / 32 / 64 / 96
  const int bm    = blockIdx.y * 128;
  const int bn    = blockIdx.x * 128;

  // staging decomposition (per thread)
  const int ar0 = tid >> 2;             // A chunk 0 row
  const int ac0 = (tid & 3) * 8;        // A chunk 0 col (halves)
  const int ar1 = (tid + 256) >> 2;     // A chunk 1 row
  const int ac1 = ac0;                  // same col pattern
  const int bk  = tid >> 3;             // B k-row (0..31)
  const int bn0 = (tid & 7) * 16;       // B n-base (0..112)

  v8f acc[4][2];
  #pragma unroll
  for (int t = 0; t < 4; ++t)
    #pragma unroll
    for (int u = 0; u < 2; ++u)
      #pragma unroll
      for (int i = 0; i < 8; ++i) acc[t][u][i] = 0.f;

  v8bf ra0, ra1, rb0, rb1;

  auto load_regs = [&](int kt) {
#if !ASYNC_A
    ra0 = *(const v8bf*)&A[(size_t)(bm + ar0) * K + (size_t)kt * 32 + ac0];
    ra1 = *(const v8bf*)&A[(size_t)(bm + ar1) * K + (size_t)kt * 32 + ac1];
#endif
    const __bf16* src = &B[(size_t)(kt * 32 + bk) * Nn + bn + bn0];
    rb0 = *(const v8bf*)src;
    rb1 = *(const v8bf*)(src + 8);
    __builtin_prefetch(src, 0, 1);  // touch next B rows into L2 (global_prefetch_b8)
  };

  auto store_stage = [&](int buf, int kt) {
#if ASYNC_A
    async_ld_b128(&A[(size_t)(bm + ar0) * K + (size_t)kt * 32 + ac0], &As[buf][ar0][ac0]);
    async_ld_b128(&A[(size_t)(bm + ar1) * K + (size_t)kt * 32 + ac1], &As[buf][ar1][ac1]);
#else
    *(v8bf*)&As[buf][ar0][ac0] = ra0;
    *(v8bf*)&As[buf][ar1][ac1] = ra1;
#endif
    #pragma unroll
    for (int i = 0; i < 8; ++i) Bs[buf][bn0 + i][bk] = rb0[i];
    #pragma unroll
    for (int i = 0; i < 8; ++i) Bs[buf][bn0 + 8 + i][bk] = rb1[i];
  };

  auto compute = [&](int buf) {
    FragBF af[4], bfr[2];
    const int c0 = (lane >> 4) * 8;     // A: K-half group per lane
    #pragma unroll
    for (int t = 0; t < 4; ++t) {
      int row = waveM + t * 16 + (lane & 15);
      af[t].h[0] = *(const v8bf*)&As[buf][row][c0];
      af[t].h[1] = *(const v8bf*)&As[buf][row][c0 + 16];
    }
    const int kb = (lane >> 4) * 8;     // B: K offset per lane half
    #pragma unroll
    for (int u = 0; u < 2; ++u) {
      int col = waveN + u * 16 + (lane & 15);
      bfr[u].h[0] = *(const v8bf*)&Bs[buf][col][kb];
      bfr[u].h[1] = *(const v8bf*)&Bs[buf][col][kb + 16];
    }
    #pragma unroll
    for (int t = 0; t < 4; ++t)
      #pragma unroll
      for (int u = 0; u < 2; ++u)
        acc[t][u] = __builtin_amdgcn_wmma_f32_16x16x32_bf16(
            false, af[t].v, false, bfr[u].v, (short)0, acc[t][u], false, false);
  };

  const int kTiles = K >> 5;

  // prologue: stage tile 0, start loads for tile 1
  load_regs(0);
  store_stage(0, 0);
  if (kTiles > 1) load_regs(1);
  wait_async_all();
  __syncthreads();

  for (int kt = 0; kt < kTiles; ++kt) {
    compute(kt & 1);
    if (kt + 1 < kTiles) {
      store_stage((kt + 1) & 1, kt + 1);      // regs hold tile kt+1 (B); A async
      if (kt + 2 < kTiles) load_regs(kt + 2); // overlap next fetch with waits
      wait_async_all();
      __syncthreads();
    }
  }

  // ---- epilogue: C/D layout lanes0-15 M=v, lanes16-31 M=v+8, N=lane&15 ----
  #pragma unroll
  for (int t = 0; t < 4; ++t) {
    #pragma unroll
    for (int u = 0; u < 2; ++u) {
      #pragma unroll
      for (int v = 0; v < 8; ++v) {
        int row = bm + waveM + t * 16 + (lane >> 4) * 8 + v;
        int col = bn + waveN + u * 16 + (lane & 15);
        size_t idx = (size_t)row * Nn + col;
        float val = acc[t][u][v];
        if (bias) val += bias[col];
        val *= scale;
        if (add0) val += add0[idx];
        if (add1) val += add1[idx];
        out[idx] = val;
      }
    }
  }
}

// ---------------------------------------------------------------------------
// Elementwise helpers
// ---------------------------------------------------------------------------
__global__ void cast_bf16_kernel(const float* __restrict__ in, __bf16* __restrict__ out, size_t n) {
  size_t i = (size_t)blockIdx.x * blockDim.x + threadIdx.x;
  if (i < n) out[i] = (__bf16)in[i];
}
__global__ void diff_bf16_kernel(const float* __restrict__ a, const float* __restrict__ b,
                                 __bf16* __restrict__ out, size_t n) {
  size_t i = (size_t)blockIdx.x * blockDim.x + threadIdx.x;
  if (i < n) out[i] = (__bf16)(a[i] - b[i]);
}
__global__ void fillf_kernel(float* p, float v, size_t n) {
  size_t i = (size_t)blockIdx.x * blockDim.x + threadIdx.x;
  if (i < n) p[i] = v;
}
__global__ void fillu_kernel(unsigned* p, unsigned v, size_t n) {
  size_t i = (size_t)blockIdx.x * blockDim.x + threadIdx.x;
  if (i < n) p[i] = v;
}
__global__ void init_bias_rows_kernel(float* u, const float* bias, int dout, size_t n) {
  size_t i = (size_t)blockIdx.x * blockDim.x + threadIdx.x;
  if (i < n) u[i] = bias[i % dout];
}

// ---------------------------------------------------------------------------
// feat5: per-edge norms / cosine / diff-norm (wave per edge)
// ---------------------------------------------------------------------------
__global__ __launch_bounds__(256)
void feat_kernel(const float* __restrict__ h, const long long* __restrict__ ei,
                 int din, float* __restrict__ feats /* NE x 4 */)
{
  int e = (int)(((size_t)blockIdx.x * blockDim.x + threadIdx.x) >> 5);
  int lane = threadIdx.x & 31;
  if (e >= NE) return;
  long long s = ei[e], d = ei[NE + e];
  const float* hs = h + (size_t)s * din;
  const float* hd = h + (size_t)d * din;
  float ss = 0.f, dd = 0.f, sd = 0.f, df = 0.f;
  for (int j = lane; j < din; j += 32) {
    float a = hs[j], b = hd[j];
    ss += a * a; dd += b * b; sd += a * b;
    float t = a - b; df += t * t;
  }
  ss = wave_sum(ss); dd = wave_sum(dd); sd = wave_sum(sd); df = wave_sum(df);
  if (lane == 0) {
    float ns = sqrtf(ss), nd = sqrtf(dd);
    feats[e * 4 + 0] = sd / (ns * nd + 1e-8f);
    feats[e * 4 + 1] = sqrtf(df) / sqrtf((float)din);
    feats[e * 4 + 2] = ns;
    feats[e * 4 + 3] = nd;
  }
}

// ---------------------------------------------------------------------------
// Edge calibration MLPs (5 -> 64 -> 1), one thread per edge
// ---------------------------------------------------------------------------
__device__ __forceinline__ float edge_mlp(const float f[5], const float* W1,
                                          const float* b1, const float* W2, float b2) {
  float y = b2;
  for (int hh = 0; hh < 64; ++hh) {
    float a = b1[hh];
    #pragma unroll
    for (int i = 0; i < 5; ++i) a += f[i] * W1[i * 64 + hh];
    y += lrelu(a) * W2[hh];
  }
  return y;
}

__global__ __launch_bounds__(256)
void edgemlp_kernel(const float* __restrict__ feats, const float* __restrict__ ew,
                    const float* spW1, const float* spb1, const float* spW2, const float* spb2,
                    const float* ftW1, const float* ftb1, const float* ftW2, const float* ftb2,
                    const float* pW1,  const float* pb1,  const float* pW2,  const float* pb2,
                    float* __restrict__ w_sp, float* __restrict__ w_ft)
{
  int e = (int)((size_t)blockIdx.x * blockDim.x + threadIdx.x);
  if (e >= NE) return;
  float w0_sp = ew[e * 2 + 0], w0_ft = ew[e * 2 + 1];
  float f[5] = { feats[e*4+0], feats[e*4+1], feats[e*4+2], feats[e*4+3], 0.f };

  f[4] = w0_sp;
  float ysp = edge_mlp(f, spW1, spb1, spW2, spb2[0]);
  float rsp = 1.f / (1.f + expf(-ysp));

  f[4] = w0_ft;
  float yft = edge_mlp(f, ftW1, ftb1, ftW2, ftb2[0]);
  float rft = 1.f / (1.f + expf(-yft));
  float yp  = edge_mlp(f, pW1, pb1, pW2, pb2[0]);
  float sp  = (yp > 20.f) ? yp : log1pf(expf(yp));

  w_sp[e] = w0_sp * rsp;
  w_ft[e] = w0_ft * rft + LAMBDA_PATCH * sp;
}

// ---------------------------------------------------------------------------
// Radix select of k-th smallest (4 passes x 8 bits, MSB first)
// ---------------------------------------------------------------------------
__global__ void sel_init_kernel(unsigned* bins, unsigned* prefix, unsigned* krem) {
  int t = threadIdx.x;
  if (t < 256) bins[t] = 0;
  if (t == 0) { *prefix = 0; *krem = KPRUNE; }
}
__global__ void sel_hist_kernel(const float* __restrict__ w, unsigned* bins,
                                const unsigned* prefix, int pass) {
  int e = (int)((size_t)blockIdx.x * blockDim.x + threadIdx.x);
  if (e >= NE) return;
  unsigned key = keyf(w[e]);
  int shift = 24 - 8 * pass;
  if (pass == 0 || (key >> (shift + 8)) == *prefix)
    atomicAdd(&bins[(key >> shift) & 255u], 1u);
}
__global__ void sel_scan_kernel(unsigned* bins, unsigned* prefix, unsigned* krem,
                                float* kth, int pass) {
  if (threadIdx.x != 0) return;
  unsigned k = *krem, cum = 0;
  int b = 0;
  for (; b < 256; ++b) { if (cum + bins[b] >= k) break; cum += bins[b]; }
  *krem = k - cum;
  unsigned pref = ((pass == 0) ? 0u : *prefix);
  pref = (pref << 8) | (unsigned)b;
  *prefix = pref;
  if (pass == 3) *kth = inv_keyf(pref);
  for (int i = 0; i < 256; ++i) bins[i] = 0;
}
__global__ void sel_threshold_kernel(float* __restrict__ w, const float* kth) {
  int e = (int)((size_t)blockIdx.x * blockDim.x + threadIdx.x);
  if (e >= NE) return;
  float t = *kth;
  float v = w[e];
  w[e] = (v >= t) ? v : 0.f;
}

// ---------------------------------------------------------------------------
// GATv2 attention: alpha_e = sum_j lrelu(xl[src]+xr[dst]+w_e*We_row) * att
// wave per edge; segment-max via monotone-key atomicMax
// ---------------------------------------------------------------------------
__global__ __launch_bounds__(256)
void alpha_kernel(const float* __restrict__ xl, const float* __restrict__ xr,
                  const long long* __restrict__ ei, const float* __restrict__ w,
                  const float* __restrict__ We, int werow,
                  const float* __restrict__ att, int dout,
                  float* __restrict__ alpha, unsigned* __restrict__ amaxk)
{
  int e = (int)(((size_t)blockIdx.x * blockDim.x + threadIdx.x) >> 5);
  int lane = threadIdx.x & 31;
  if (e >= NE) return;
  long long s = ei[e], d = ei[NE + e];
  float we_ = w[e];
  const float* pl = xl + (size_t)s * dout;
  const float* pr = xr + (size_t)d * dout;
  const float* pe = We + (size_t)werow * dout;
  float acc = 0.f;
  for (int j = lane; j < dout; j += 32)
    acc += lrelu(pl[j] + pr[j] + we_ * pe[j]) * att[j];
  acc = wave_sum(acc);
  if (lane == 0) {
    alpha[e] = acc;
    atomicMax(&amaxk[d], keyf(acc));
  }
}

__global__ void amax_fin_kernel(const unsigned* __restrict__ amaxk,
                                float* __restrict__ amaxf, int n) {
  int i = (int)((size_t)blockIdx.x * blockDim.x + threadIdx.x);
  if (i >= n) return;
  unsigned k = amaxk[i];
  amaxf[i] = (k == KEY_NEG_INF) ? 0.f : inv_keyf(k);
}

__global__ void exden_kernel(const float* __restrict__ alpha, const float* __restrict__ amaxf,
                             const long long* __restrict__ ei,
                             float* __restrict__ ex, float* __restrict__ den) {
  int e = (int)((size_t)blockIdx.x * blockDim.x + threadIdx.x);
  if (e >= NE) return;
  long long d = ei[NE + e];
  float x = expf(alpha[e] - amaxf[d]);
  ex[e] = x;
  atomicAdd(&den[d], x);
}

__global__ __launch_bounds__(256)
void scatter_kernel(const float* __restrict__ ex, const float* __restrict__ den,
                    const float* __restrict__ xl, const long long* __restrict__ ei,
                    int dout, float* __restrict__ u)
{
  int e = (int)(((size_t)blockIdx.x * blockDim.x + threadIdx.x) >> 5);
  int lane = threadIdx.x & 31;
  if (e >= NE) return;
  long long s = ei[e], d = ei[NE + e];
  float coef = ex[e] / den[d];
  const float* src = xl + (size_t)s * dout;
  float* dst = u + (size_t)d * dout;
  for (int j = lane; j < dout; j += 32)
    atomicAdd(&dst[j], coef * src[j]);
}

// ---------------------------------------------------------------------------
// LayerNorm (over features) + LeakyReLU; emits f32 + bf16 activations
// ---------------------------------------------------------------------------
__global__ __launch_bounds__(256)
void ln_lrelu_kernel(const float* __restrict__ x, const float* __restrict__ g,
                     const float* __restrict__ b, int dout,
                     float* __restrict__ outf, __bf16* __restrict__ outbf)
{
  __shared__ float red[16];
  int row = blockIdx.x, tid = threadIdx.x;
  const float* xr = x + (size_t)row * dout;
  float s = 0.f, s2 = 0.f;
  for (int j = tid; j < dout; j += 256) { float v = xr[j]; s += v; s2 += v * v; }
  s = wave_sum(s); s2 = wave_sum(s2);
  int lane = tid & 31, w = tid >> 5;
  if (lane == 0) { red[w] = s; red[8 + w] = s2; }
  __syncthreads();
  if (w == 0) {
    s  = (lane < 8) ? red[lane] : 0.f;
    s2 = (lane < 8) ? red[8 + lane] : 0.f;
    #pragma unroll
    for (int o = 4; o; o >>= 1) { s += __shfl_xor(s, o, 32); s2 += __shfl_xor(s2, o, 32); }
    if (lane == 0) { red[0] = s; red[8] = s2; }
  }
  __syncthreads();
  float mean = red[0] / dout;
  float var  = red[8] / dout - mean * mean;
  float rstd = rsqrtf(var + 1e-5f);
  for (int j = tid; j < dout; j += 256) {
    float v = (xr[j] - mean) * rstd * g[j] + b[j];
    v = lrelu(v);
    outf[(size_t)row * dout + j] = v;
    outbf[(size_t)row * dout + j] = (__bf16)v;
  }
}

// ---------------------------------------------------------------------------
// Host orchestration
// ---------------------------------------------------------------------------
static inline void gemm(const __bf16* A, const __bf16* B, const float* bias,
                        const float* add0, const float* add1, float scale,
                        float* out, int M, int K, int Nn, hipStream_t s) {
  dim3 grid(Nn / 128, M / 128);
  gemm_bf16_kernel<<<grid, 256, 0, s>>>(A, B, bias, add0, add1, scale, out, M, K, Nn);
}

extern "C" void kernel_launch(void* const* d_in, const int* in_sizes, int n_in,
                              void* d_out, int out_size, void* d_ws, size_t ws_size,
                              hipStream_t stream)
{
  (void)in_sizes; (void)n_in; (void)out_size; (void)ws_size;
  const float*      x  = (const float*)d_in[0];
  const long long*  ei = (const long long*)d_in[1];
  const float*      ew = (const float*)d_in[2];

  // params: pytree-flattened (sorted keys), 34 arrays per layer, base 3
  const int PBASE = 3, PL = 34;
  auto P = [&](int L, int off) -> const float* {
    return (const float*)d_in[PBASE + L * PL + off];
  };
  // per-layer offsets (sorted dict keys):
  // calib_ft: W1=0 W2=1 b1=2 b2=3 | calib_sp: 4..7
  // conv_ft:  We=8 Wl=9 Wr=10 att=11 bias=12 bl=13 br=14 | conv_sp: 15..21
  // cross_ft: b=22 w=23 | cross_sp: b=24 w=25 | norm: b=26 g=27
  // patch_ft: W1=28 W2=29 b1=30 b2=31 | skip: b=32 w=33

  // ---- workspace carve ----
  char* wp = (char*)d_ws;
  auto alloc = [&](size_t bytes) -> void* {
    void* p = (void*)wp; wp += (bytes + 255) & ~(size_t)255; return p;
  };
  const size_t NDMAX = (size_t)NN * 2048;
  __bf16* hbf   = (__bf16*)alloc(NDMAX * 2);
  float*  hA    = (float*) alloc(NDMAX * 4);
  float*  hB    = (float*) alloc(NDMAX * 4);
  float*  xl    = (float*) alloc(NDMAX * 4);
  float*  xr    = (float*) alloc(NDMAX * 4);
  float*  usp   = (float*) alloc(NDMAX * 4);
  float*  uft   = (float*) alloc(NDMAX * 4);
  __bf16* dbf   = (__bf16*)alloc(NDMAX * 2);
  __bf16* wbf   = (__bf16*)alloc((size_t)2048 * 2048 * 2);
  float*  feats = (float*) alloc((size_t)NE * 4 * 4);
  float*  wsp   = (float*) alloc((size_t)NE * 4);
  float*  wft   = (float*) alloc((size_t)NE * 4);
  float*  alpha = (float*) alloc((size_t)NE * 4);
  float*  exb   = (float*) alloc((size_t)NE * 4);
  unsigned* amaxk = (unsigned*)alloc((size_t)NN * 4);
  float*  amaxf = (float*) alloc((size_t)NN * 4);
  float*  den   = (float*) alloc((size_t)NN * 4);
  unsigned* bins   = (unsigned*)alloc(256 * 4);
  unsigned* prefix = (unsigned*)alloc(4);
  unsigned* krem   = (unsigned*)alloc(4);
  float*    kth    = (float*)   alloc(4);

  const int dims[3] = {2048, 1024, 512};
  const int EB = NE / 256;            // thread-per-edge blocks
  const int EWB = (NE * 32) / 256;    // wave-per-edge blocks

  // initial bf16 activation
  {
    size_t n = (size_t)NN * 512;
    cast_bf16_kernel<<<(unsigned)((n + 255) / 256), 256, 0, stream>>>(x, hbf, n);
  }

  const float* hf = x;
  int din = 512;
  for (int L = 0; L < 3; ++L) {
    int dout = dims[L];
    size_t nd = (size_t)NN * dout;
    unsigned ndB = (unsigned)((nd + 255) / 256);

    // edge features + calibration
    feat_kernel<<<EWB, 256, 0, stream>>>(hf, ei, din, feats);
    edgemlp_kernel<<<EB, 256, 0, stream>>>(feats, ew,
        P(L,4), P(L,6), P(L,5), P(L,7),        // calib_sp
        P(L,0), P(L,2), P(L,1), P(L,3),        // calib_ft
        P(L,28), P(L,30), P(L,29), P(L,31),    // patch_ft
        wsp, wft);

    // prune w_sp at k-th smallest
    sel_init_kernel<<<1, 256, 0, stream>>>(bins, prefix, krem);
    for (int p = 0; p < 4; ++p) {
      sel_hist_kernel<<<EB, 256, 0, stream>>>(wsp, bins, prefix, p);
      sel_scan_kernel<<<1, 1, 0, stream>>>(bins, prefix, krem, kth, p);
    }
    sel_threshold_kernel<<<EB, 256, 0, stream>>>(wsp, kth);

    // two GAT branches: mode 0 = sp (We row 0), mode 1 = ft (We row 1)
    for (int mode = 0; mode < 2; ++mode) {
      int gb = mode ? 8 : 15;  // conv_ft base 8, conv_sp base 15
      const float* Wl  = P(L, gb + 1);
      const float* Wr  = P(L, gb + 2);
      const float* att = P(L, gb + 3);
      const float* bia = P(L, gb + 4);
      const float* bl  = P(L, gb + 5);
      const float* br  = P(L, gb + 6);
      const float* We  = P(L, gb + 0);
      const float* wE  = mode ? wft : wsp;
      float* u = mode ? uft : usp;

      size_t nw = (size_t)din * dout;
      unsigned nwB = (unsigned)((nw + 255) / 256);
      cast_bf16_kernel<<<nwB, 256, 0, stream>>>(Wl, wbf, nw);
      gemm(hbf, wbf, bl, nullptr, nullptr, 1.f, xl, NN, din, dout, stream);
      cast_bf16_kernel<<<nwB, 256, 0, stream>>>(Wr, wbf, nw);
      gemm(hbf, wbf, br, nullptr, nullptr, 1.f, xr, NN, din, dout, stream);

      fillu_kernel<<<(NN + 255) / 256, 256, 0, stream>>>(amaxk, KEY_NEG_INF, (size_t)NN);
      fillf_kernel<<<(NN + 255) / 256, 256, 0, stream>>>(den, 1e-16f, (size_t)NN);
      init_bias_rows_kernel<<<ndB, 256, 0, stream>>>(u, bia, dout, nd);

      alpha_kernel<<<EWB, 256, 0, stream>>>(xl, xr, ei, wE, We, mode, att, dout, alpha, amaxk);
      amax_fin_kernel<<<(NN + 255) / 256, 256, 0, stream>>>(amaxk, amaxf, NN);
      exden_kernel<<<EB, 256, 0, stream>>>(alpha, amaxf, ei, exb, den);
      scatter_kernel<<<EWB, 256, 0, stream>>>(exb, den, xl, ei, dout, u);
    }

    // cross coupling (sequential: u_ft update sees updated u_sp)
    size_t nc = (size_t)dout * dout;
    unsigned ncB = (unsigned)((nc + 255) / 256);
    diff_bf16_kernel<<<ndB, 256, 0, stream>>>(uft, usp, dbf, nd);
    cast_bf16_kernel<<<ncB, 256, 0, stream>>>(P(L, 25), wbf, nc);      // cross_sp.w
    gemm(dbf, wbf, P(L, 24), usp, nullptr, BETA_CROSS, usp, NN, dout, dout, stream);
    diff_bf16_kernel<<<ndB, 256, 0, stream>>>(usp, uft, dbf, nd);
    cast_bf16_kernel<<<ncB, 256, 0, stream>>>(P(L, 23), wbf, nc);      // cross_ft.w
    gemm(dbf, wbf, P(L, 22), uft, nullptr, BETA_CROSS, uft, NN, dout, dout, stream);

    // skip + residual sum -> pre-LN (reuse xl)
    size_t nw = (size_t)din * dout;
    cast_bf16_kernel<<<(unsigned)((nw + 255) / 256), 256, 0, stream>>>(P(L, 33), wbf, nw);
    gemm(hbf, wbf, P(L, 32), usp, uft, 1.f, xl, NN, din, dout, stream);

    // layernorm + lrelu -> next activation (f32 + bf16)
    float* hnext = (L == 2) ? (float*)d_out : ((L & 1) ? hB : hA);
    ln_lrelu_kernel<<<NN, 256, 0, stream>>>(xl, P(L, 27), P(L, 26), dout, hnext, hbf);

    hf = hnext;
    din = dout;
  }
}
